// ProtNet_5566277616160
// MI455X (gfx1250) — matrix operand order repaired and checked by reading, not compiled, not run on hardware
//
#include <hip/hip_runtime.h>
#include <math.h>

// ---------------- model constants ----------------
#define BBATCH 8
#define NRMAX  768
#define NLMAX  384
#define RESC   21
#define DIMC   128
#define NHEAD  4
#define DHEAD  16
#define NDEG   3
#define KNBR   10
#define TDEPTH 4
#define PDEPTH 4

typedef _Float16 half_t;
typedef _Float16 v16h __attribute__((ext_vector_type(16)));
typedef float    v8f  __attribute__((ext_vector_type(8)));

struct __attribute__((aligned(16))) U4 { unsigned int x, y, z, w; };

// =====================================================================
// WMMA GEMM:  Y[M,Nout] (f32) = X[M,Kd] (f16, row-major) * Wt[Nout,Kd]^T
// grid = (M/128, Nout/64), block = 256 (8 wave32).
// Each wave owns a 16x64 output strip: the A fragment (2 global b128
// loads) is reused across 4 v_wmma_f32_16x16x32_f16 issues; B fragments
// come from a 64-row Wt tile staged in LDS (<=48KB @ K=384).
// =====================================================================
__global__ __launch_bounds__(256)
void wmma_gemm_f16(const half_t* __restrict__ X, const half_t* __restrict__ Wt,
                   float* __restrict__ Y, int M, int Kd, int Nout)
{
    extern __shared__ char smem[];
    half_t* ldsW = (half_t*)smem;
    const int tid  = threadIdx.x;
    const int wave = tid >> 5;
    const int lane = tid & 31;
    const int n0   = blockIdx.y * 64;

    // stage 64 rows of Wt (B operand tile) into LDS, 128b chunks
    {
        const U4* src = (const U4*)(Wt + (long)n0 * Kd);
        U4*       dst = (U4*)ldsW;
        const int cnt = (64 * Kd) / 8;        // U4 units
        for (int i = tid; i < cnt; i += 256) dst[i] = src[i];
    }
    __syncthreads();

    const int  mTile = blockIdx.x * 8 + wave; // 16-row tile per wave
    const int  nlane = lane & 15;
    const int  kh    = lane >> 4;
    const long rowA  = (long)(mTile * 16 + nlane) * Kd;

    v8f acc0 = {}, acc1 = {}, acc2 = {}, acc3 = {};
    for (int k0 = 0; k0 < Kd; k0 += 32) {
        union { v16h v; U4 q[2]; } A, B0, B1, B2, B3;
        // A fragment: lane holds row M=nlane, K in [k0+8kh, +8) U [k0+16+8kh, +8)
        A.q[0] = *(const U4*)(X + rowA + k0 + 8 * kh);
        A.q[1] = *(const U4*)(X + rowA + k0 + 16 + 8 * kh);
        // B fragments from LDS: lane holds col N=nlane(+16t), K in [k0+8kh, +8) U [k0+16+8kh, +8)
        const int boff = k0 + 8 * kh;
        const U4* bp0 = (const U4*)(ldsW + (long)(nlane     ) * Kd + boff);
        const U4* bp1 = (const U4*)(ldsW + (long)(nlane + 16) * Kd + boff);
        const U4* bp2 = (const U4*)(ldsW + (long)(nlane + 32) * Kd + boff);
        const U4* bp3 = (const U4*)(ldsW + (long)(nlane + 48) * Kd + boff);
        B0.q[0] = bp0[0]; B0.q[1] = bp0[2];
        B1.q[0] = bp1[0]; B1.q[1] = bp1[2];
        B2.q[0] = bp2[0]; B2.q[1] = bp2[2];
        B3.q[0] = bp3[0]; B3.q[1] = bp3[2];
        if (k0 + 32 < Kd) __builtin_prefetch(X + rowA + k0 + 96, 0, 1);
        acc0 = __builtin_amdgcn_wmma_f32_16x16x32_f16(false, A.v, false, B0.v,
                                                      (short)0, acc0, false, false);
        acc1 = __builtin_amdgcn_wmma_f32_16x16x32_f16(false, A.v, false, B1.v,
                                                      (short)0, acc1, false, false);
        acc2 = __builtin_amdgcn_wmma_f32_16x16x32_f16(false, A.v, false, B2.v,
                                                      (short)0, acc2, false, false);
        acc3 = __builtin_amdgcn_wmma_f32_16x16x32_f16(false, A.v, false, B3.v,
                                                      (short)0, acc3, false, false);
    }
    // C/D layout: VGPR r -> row (r + 8*kh), lane n -> col n
    const int mBase = mTile * 16 + 8 * kh;
    {
        float* yp = Y + (long)mBase * Nout + n0 + nlane;
#pragma unroll
        for (int r = 0; r < 8; ++r) yp[(long)r * Nout] = acc0[r];
        yp += 16;
#pragma unroll
        for (int r = 0; r < 8; ++r) yp[(long)r * Nout] = acc1[r];
        yp += 16;
#pragma unroll
        for (int r = 0; r < 8; ++r) yp[(long)r * Nout] = acc2[r];
        yp += 16;
#pragma unroll
        for (int r = 0; r < 8; ++r) yp[(long)r * Nout] = acc3[r];
    }
}

// =====================================================================
// small conversion / packing kernels
// =====================================================================
__global__ void transpose_cvt_k(const float* __restrict__ W, half_t* __restrict__ Wt,
                                int Kd, int Nout)
{
    long i = (long)blockIdx.x * blockDim.x + threadIdx.x;
    if (i >= (long)Kd * Nout) return;
    int k = (int)(i / Nout), n = (int)(i % Nout);
    Wt[(long)n * Kd + k] = (half_t)W[i];
}

__global__ void pack_conv_w_k(const float* __restrict__ w, half_t* __restrict__ Wt,
                              int Cin, int Kdp)
{
    long i = (long)blockIdx.x * blockDim.x + threadIdx.x;
    if (i >= (long)DIMC * Kdp) return;
    int o = (int)(i / Kdp), col = (int)(i % Kdp);
    float v = 0.f;
    if (col < 3 * Cin) { int t = col / Cin, c = col % Cin; v = w[((long)o * Cin + c) * 3 + t]; }
    Wt[i] = (half_t)v;
}

__global__ void pack_im2col_k(const float* __restrict__ x, half_t* __restrict__ X,
                              int N, int C, int Kdp)
{
    long i = (long)blockIdx.x * blockDim.x + threadIdx.x;
    long total = (long)BBATCH * N * Kdp;
    if (i >= total) return;
    int  col = (int)(i % Kdp);
    long rn  = i / Kdp;
    int  n   = (int)(rn % N);
    int  b   = (int)(rn / N);
    float v = 0.f;
    if (col < 3 * C) {
        int t = col / C, c = col % C, ns = n + t - 1;
        if (ns >= 0 && ns < N) v = x[((long)b * N + ns) * C + c];
    }
    X[i] = (half_t)v;
}

__global__ void bias_silu_k(const float* __restrict__ y, const float* __restrict__ bias,
                            const float* __restrict__ res, float* __restrict__ out, long rows)
{
    long i = (long)blockIdx.x * blockDim.x + threadIdx.x;
    if (i >= rows * DIMC) return;
    int c = (int)(i % DIMC);
    float v = y[i] + bias[c];
    v = v / (1.f + __expf(-v));              // silu
    out[i] = (res ? res[i] : 0.f) + v;
}

__global__ void vec_proj_k(const float* __restrict__ ang, const float* __restrict__ w,
                           float* __restrict__ f1, long rows)
{
    long i = (long)blockIdx.x * blockDim.x + threadIdx.x;
    if (i >= rows * DIMC) return;
    int  o = (int)(i % DIMC);
    long r = i / DIMC;
#pragma unroll
    for (int m = 0; m < 3; ++m) {
        float acc = 0.f;
#pragma unroll
        for (int c = 0; c < 3; ++c) acc += ang[r * 9 + c * 3 + m] * w[c * DIMC + o];
        f1[i * 3 + m] = acc;
    }
}

__global__ void invariants_k(const float* __restrict__ f0, const float* __restrict__ f1,
                             const float* __restrict__ f2, half_t* __restrict__ invb,
                             long rows, int ndeg, int F)
{
    long i = (long)blockIdx.x * blockDim.x + threadIdx.x;
    if (i >= rows * DIMC) return;
    int  c = (int)(i % DIMC);
    long r = i / DIMC;
    float a0 = f0[i];
    invb[r * F + c] = (half_t)sqrtf(a0 * a0 + 1e-8f);
    float s1 = 1e-8f;
#pragma unroll
    for (int m = 0; m < 3; ++m) { float v = f1[i * 3 + m]; s1 += v * v; }
    invb[r * F + DIMC + c] = (half_t)sqrtf(s1);
    if (ndeg == 3) {
        float s2 = 1e-8f;
#pragma unroll
        for (int m = 0; m < 5; ++m) { float v = f2[i * 5 + m]; s2 += v * v; }
        invb[r * F + 2 * DIMC + c] = (half_t)sqrtf(s2);
    }
}

__global__ void slice_pack_k(const float* __restrict__ f, half_t* __restrict__ X,
                             long rows, int Mm, int m)
{
    long i = (long)blockIdx.x * blockDim.x + threadIdx.x;
    if (i >= rows * DIMC) return;
    X[i] = (half_t)f[i * Mm + m];
}

__global__ void slice_add_k(const float* __restrict__ y, float* __restrict__ f,
                            long rows, int Mm, int m)
{
    long i = (long)blockIdx.x * blockDim.x + threadIdx.x;
    if (i >= rows * DIMC) return;
    f[i * Mm + m] += y[i];
}

__global__ void zero_k(float* p, long n)
{
    long i = (long)blockIdx.x * blockDim.x + threadIdx.x;
    if (i < n) p[i] = 0.f;
}

// =====================================================================
// sparse neighbor top-K with chain-adjacency bonus
// =====================================================================
__global__ void topk_k(const float* __restrict__ pos, const unsigned char* __restrict__ msk,
                       int* __restrict__ nidx, unsigned char* __restrict__ nmk, int N)
{
    int b = blockIdx.y;
    int n = blockIdx.x * blockDim.x + threadIdx.x;
    if (n >= N) return;
    long base = (long)b * N;
    float px = pos[(base + n) * 3 + 0], py = pos[(base + n) * 3 + 1], pz = pos[(base + n) * 3 + 2];
    float bs[KNBR]; int bi[KNBR]; unsigned char bm[KNBR];
#pragma unroll
    for (int t = 0; t < KNBR; ++t) { bs[t] = -3e38f; bi[t] = 0; bm[t] = 0; }
    for (int j = 0; j < N; ++j) {
        if (j == n) continue;
        int vj = msk[base + j] ? 1 : 0;
        float dx = pos[(base + j) * 3 + 0] - px;
        float dy = pos[(base + j) * 3 + 1] - py;
        float dz = pos[(base + j) * 3 + 2] - pz;
        float sc = -(dx * dx + dy * dy + dz * dz);
        if (j == n + 1 && msk[base + n]) sc += 1e6f;
        else if (j + 1 == n && msk[base + j]) sc += 1e6f;
        if (!vj) sc = -3e38f;
        if (sc > bs[KNBR - 1]) {
            int p = KNBR - 1;
            while (p > 0 && sc > bs[p - 1]) {
                bs[p] = bs[p - 1]; bi[p] = bi[p - 1]; bm[p] = bm[p - 1]; --p;
            }
            bs[p] = sc; bi[p] = j; bm[p] = (unsigned char)vj;
        }
    }
#pragma unroll
    for (int t = 0; t < KNBR; ++t) {
        nidx[(base + n) * KNBR + t] = bi[t];
        nmk [(base + n) * KNBR + t] = bm[t];
    }
}

// =====================================================================
// attention over K neighbors + SH messages + neighbor aggregation
// one block (128 threads = channel) per node
// =====================================================================
__global__ __launch_bounds__(128)
void attn_msg_k(const float* __restrict__ q,  const float* __restrict__ kk,
                const float* __restrict__ v0, const float* __restrict__ v1,
                const float* __restrict__ v2,
                const float* __restrict__ f0, const float* __restrict__ f1,
                const float* __restrict__ f2,
                const float* __restrict__ pos, const int* __restrict__ nidx,
                const unsigned char* __restrict__ nmk,
                float* __restrict__ o0, float* __restrict__ o1, float* __restrict__ o2,
                float* __restrict__ a0, float* __restrict__ a1, float* __restrict__ a2,
                int N, int ndeg)
{
    const long r   = blockIdx.x;
    const int  tid = threadIdx.x;
    const int  b   = (int)(r / N);

    __shared__ int   sIdx[KNBR];
    __shared__ float sMsk[KNBR];
    __shared__ float sLog[NHEAD][KNBR];
    __shared__ float sMx[NHEAD], sDen[NHEAD];
    __shared__ float sAttn[KNBR];
    __shared__ float sY[KNBR][9];

    if (tid < KNBR) {
        sIdx[tid] = nidx[r * KNBR + tid];
        sMsk[tid] = nmk[r * KNBR + tid] ? 1.f : 0.f;
    }
    __syncthreads();

    if (tid < NHEAD * KNBR) {
        int h = tid / KNBR, k = tid % KNBR;
        long j = (long)b * N + sIdx[k];
        float acc = 0.f;
#pragma unroll
        for (int d = 0; d < DHEAD; ++d)
            acc += q[r * 64 + h * DHEAD + d] * kk[j * 64 + h * DHEAD + d];
        sLog[h][k] = (sMsk[k] > 0.f) ? acc * 0.25f : -1e9f;   // 1/sqrt(16)
    }
    __syncthreads();
    if (tid < NHEAD) {
        float mx = -3e38f;
        for (int k = 0; k < KNBR; ++k) mx = fmaxf(mx, sLog[tid][k]);
        float den = 0.f;
        for (int k = 0; k < KNBR; ++k) den += __expf(sLog[tid][k] - mx);
        sMx[tid] = mx; sDen[tid] = den;
    }
    __syncthreads();
    if (tid < KNBR) {
        float s = 0.f;
        for (int h = 0; h < NHEAD; ++h) s += __expf(sLog[h][tid] - sMx[h]) / sDen[h];
        sAttn[tid] = (s / (float)NHEAD) * sMsk[tid];
        long j = (long)b * N + sIdx[tid];
        float rx = pos[j * 3 + 0] - pos[r * 3 + 0];
        float ry = pos[j * 3 + 1] - pos[r * 3 + 1];
        float rz = pos[j * 3 + 2] - pos[r * 3 + 2];
        float nrm = sqrtf(rx * rx + ry * ry + rz * rz) + 1e-8f;
        float x = rx / nrm, y = ry / nrm, z = rz / nrm;
        sY[tid][0] = 1.f;
        sY[tid][1] = x;  sY[tid][2] = y;  sY[tid][3] = z;
        sY[tid][4] = x * y; sY[tid][5] = y * z; sY[tid][6] = 3.f * z * z - 1.f;
        sY[tid][7] = x * z; sY[tid][8] = x * x - y * y;
    }
    __syncthreads();

    const int c = tid;
    float m0 = 0.f, m1[3] = {0, 0, 0}, m2[5] = {0, 0, 0, 0, 0};
    float g0 = 0.f, g1[3] = {0, 0, 0}, g2[5] = {0, 0, 0, 0, 0};
    for (int k = 0; k < KNBR; ++k) {
        float a = sAttn[k];
        long  j = ((long)b * N + sIdx[k]) * DIMC + c;
        float av0 = a * v0[j], av1 = a * v1[j], av2 = a * v2[j];
        m0 += av0 * sY[k][0];
#pragma unroll
        for (int m = 0; m < 3; ++m) m1[m] += av1 * sY[k][1 + m];
#pragma unroll
        for (int m = 0; m < 5; ++m) m2[m] += av2 * sY[k][4 + m];
        g0 += a * f0[j];
#pragma unroll
        for (int m = 0; m < 3; ++m) g1[m] += a * f1[j * 3 + m];
        if (ndeg == 3) {
#pragma unroll
            for (int m = 0; m < 5; ++m) g2[m] += a * f2[j * 5 + m];
        }
    }
    long i = r * DIMC + c;
    o0[i] = f0[i] + m0;
#pragma unroll
    for (int m = 0; m < 3; ++m) o1[i * 3 + m] = f1[i * 3 + m] + m1[m];
    if (ndeg == 3) {
#pragma unroll
        for (int m = 0; m < 5; ++m) o2[i * 5 + m] = f2[i * 5 + m] + m2[m];
#pragma unroll
        for (int m = 0; m < 5; ++m) a2[i * 5 + m] = g2[m];
    } else {
#pragma unroll
        for (int m = 0; m < 5; ++m) o2[i * 5 + m] = m2[m];
    }
    a0[i] = g0;
#pragma unroll
    for (int m = 0; m < 3; ++m) a1[i * 3 + m] = g1[m];
}

// =====================================================================
// pooling
// =====================================================================
__global__ void gate_k(const float* __restrict__ f0, const unsigned char* __restrict__ msk,
                       const float* __restrict__ pw, const float* __restrict__ pb,
                       float* __restrict__ g, float* __restrict__ gsum, int N, long rows)
{
    long i = (long)blockIdx.x * blockDim.x + threadIdx.x;
    if (i >= rows) return;
    float acc = pb[0];
    for (int c = 0; c < DIMC; ++c) acc += f0[i * DIMC + c] * pw[c];
    float gv = 1.f / (1.f + __expf(-acc));
    gv *= msk[i] ? 1.f : 0.f;
    g[i] = gv;
    atomicAdd(&gsum[i / N], gv);
}

__global__ void pool_accum_k(const float* __restrict__ f0, const float* __restrict__ f1,
                             const float* __restrict__ f2, const float* __restrict__ g,
                             float* __restrict__ praw, int N)
{
    long i = (long)blockIdx.x * blockDim.x + threadIdx.x;
    if (i >= (long)BBATCH * DIMC * 9) return;
    int mt = (int)(i % 9);
    int c  = (int)((i / 9) % DIMC);
    int b  = (int)(i / (9 * DIMC));
    float acc = 0.f;
    for (int n = 0; n < N; ++n) {
        long r  = (long)b * N + n;
        float gv = g[r];
        float fv = (mt == 0) ? f0[r * DIMC + c]
                 : (mt < 4)  ? f1[(r * DIMC + c) * 3 + (mt - 1)]
                             : f2[(r * DIMC + c) * 5 + (mt - 4)];
        acc += gv * fv;
    }
    praw[(long)(b * DIMC + c) * 9 + mt] = acc;
}

__global__ void pool_lin_k(const float* __restrict__ praw, const float* __restrict__ gsum,
                           const float* __restrict__ l0, const float* __restrict__ l1,
                           const float* __restrict__ l2,
                           float* __restrict__ p0, float* __restrict__ p1,
                           float* __restrict__ p2, int slot)
{
    long i = (long)blockIdx.x * blockDim.x + threadIdx.x;
    if (i >= (long)BBATCH * DIMC * 9) return;
    int mt = (int)(i % 9);
    int o  = (int)((i / 9) % DIMC);
    int b  = (int)(i / (9 * DIMC));
    const float* lw = (mt == 0) ? l0 : (mt < 4) ? l1 : l2;
    float acc = 0.f;
    for (int c = 0; c < DIMC; ++c) acc += praw[(long)(b * DIMC + c) * 9 + mt] * lw[c * DIMC + o];
    acc /= gsum[b];
    if (mt == 0)      p0[((long)(b * 2 + slot) * DIMC + o)]                     = acc;
    else if (mt < 4)  p1[((long)(b * 2 + slot) * DIMC + o) * 3 + (mt - 1)]      = acc;
    else              p2[((long)(b * 2 + slot) * DIMC + o) * 5 + (mt - 4)]      = acc;
}

__global__ void pos2_k(const float* __restrict__ pos, const unsigned char* __restrict__ msk,
                       float* __restrict__ pos2, int slot, int N)
{
    int b = blockIdx.x;
    if (threadIdx.x != 0) return;
    float sx = 0.f, sy = 0.f, sz = 0.f, sm = 0.f;
    for (int n = 0; n < N; ++n) {
        long r = (long)b * N + n;
        float m = msk[r] ? 1.f : 0.f;
        sx += pos[r * 3 + 0] * m; sy += pos[r * 3 + 1] * m; sz += pos[r * 3 + 2] * m;
        sm += m;
    }
    pos2[(b * 2 + slot) * 3 + 0] = sx / sm;
    pos2[(b * 2 + slot) * 3 + 1] = sy / sm;
    pos2[(b * 2 + slot) * 3 + 2] = sz / sm;
}

// =====================================================================
// fused tail: ds-linears + time-emb + PDEPTH ptrans layers + output head
// (softmax over K=1 neighbor is identically 1, so wq/wk drop out exactly)
// =====================================================================
struct FinalP {
    const float* ds0; const float* ds1; const float* ds2;
    const float* pout1; const float* fw;
    const float* wv[PDEPTH][3];
    const float* ws[PDEPTH][3];
    const float* wn[PDEPTH][3];
};

__global__ __launch_bounds__(256)
void final_k(FinalP P, const float* __restrict__ pool0, const float* __restrict__ pool1,
             const float* __restrict__ pool2, const float* __restrict__ pos2,
             const float* __restrict__ tt, float* __restrict__ outp)
{
    __shared__ float fa[2][DIMC][9];
    __shared__ float fb[2][DIMC][9];
    __shared__ float sInv[2][3 * DIMC];
    __shared__ float sV[3][2][DIMC];
    __shared__ float sUY[2][9];
    __shared__ float sT2[2][DIMC][3];

    int b = blockIdx.x, tid = threadIdx.x;
    for (int i = tid; i < 2 * DIMC * 9; i += 256) {
        int s = i / (DIMC * 9), c = (i / 9) % DIMC, m = i % 9;
        fa[s][c][m] = 0.f; fb[s][c][m] = 0.f;
    }
    __syncthreads();

    // ds linears per degree
    for (int i = tid; i < 2 * 64; i += 256) {
        int s = i / 64, o = i % 64;
        float acc = 0.f;
        for (int c = 0; c < DIMC; ++c) acc += pool0[(long)(b * 2 + s) * DIMC + c] * P.ds0[c * 64 + o];
        fa[s][o][0] = acc;
    }
    for (int i = tid; i < 2 * DIMC * 3; i += 256) {
        int s = i / (DIMC * 3), o = (i / 3) % DIMC, m = i % 3;
        float acc = 0.f;
        for (int c = 0; c < DIMC; ++c)
            acc += pool1[((long)(b * 2 + s) * DIMC + c) * 3 + m] * P.ds1[c * DIMC + o];
        fa[s][o][1 + m] = acc;
    }
    for (int i = tid; i < 2 * DIMC * 5; i += 256) {
        int s = i / (DIMC * 5), o = (i / 5) % DIMC, m = i % 5;
        float acc = 0.f;
        for (int c = 0; c < DIMC; ++c)
            acc += pool2[((long)(b * 2 + s) * DIMC + c) * 5 + m] * P.ds2[c * DIMC + o];
        fa[s][o][4 + m] = acc;
    }
    // time embedding appended along deg-0 channel axis
    for (int i = tid; i < 2 * 64; i += 256) {
        int s = i / 64, j = i % 64, jj = j % 32;
        float f = __expf((float)jj * (-9.210340371976184f / 31.f));
        float e = tt[b] * f;
        fa[s][64 + j][0] = (j < 32) ? __sinf(e) : __cosf(e);
    }
    // SH basis of the single neighbor direction per slot
    if (tid < 2) {
        int s = tid, o = 1 - s;
        float rx = pos2[(b * 2 + o) * 3 + 0] - pos2[(b * 2 + s) * 3 + 0];
        float ry = pos2[(b * 2 + o) * 3 + 1] - pos2[(b * 2 + s) * 3 + 1];
        float rz = pos2[(b * 2 + o) * 3 + 2] - pos2[(b * 2 + s) * 3 + 2];
        float nrm = sqrtf(rx * rx + ry * ry + rz * rz) + 1e-8f;
        float x = rx / nrm, y = ry / nrm, z = rz / nrm;
        sUY[s][0] = 1.f; sUY[s][1] = x; sUY[s][2] = y; sUY[s][3] = z;
        sUY[s][4] = x * y; sUY[s][5] = y * z; sUY[s][6] = 3.f * z * z - 1.f;
        sUY[s][7] = x * z; sUY[s][8] = x * x - y * y;
    }
    __syncthreads();

    float (*cur)[DIMC][9] = fa;
    float (*nxt)[DIMC][9] = fb;
    for (int L = 0; L < PDEPTH; ++L) {
        for (int i = tid; i < 2 * 3 * DIMC; i += 256) {
            int s = i / (3 * DIMC), f = i % (3 * DIMC), d = f / DIMC, c = f % DIMC;
            int mo = (d == 0) ? 0 : ((d == 1) ? 1 : 4);
            int mm = 2 * d + 1;
            float acc = 1e-8f;
            for (int m = 0; m < mm; ++m) { float v = cur[s][c][mo + m]; acc += v * v; }
            sInv[s][f] = sqrtf(acc);
        }
        __syncthreads();
        for (int i = tid; i < 3 * 2 * DIMC; i += 256) {
            int d = i / (2 * DIMC), s = (i / DIMC) % 2, c = i % DIMC;
            const float* w = P.wv[L][d];
            float acc = 0.f;
            for (int f = 0; f < 3 * DIMC; ++f) acc += sInv[s][f] * w[f * DIMC + c];
            sV[d][s][c] = acc;
        }
        __syncthreads();
        for (int i = tid; i < 2 * DIMC * 9; i += 256) {
            int s = i / (DIMC * 9), c = (i / 9) % DIMC, mt = i % 9;
            int d = (mt == 0) ? 0 : ((mt < 4) ? 1 : 2);
            int o = 1 - s;
            const float* wsd = P.ws[L][d];
            const float* wnd = P.wn[L][d];
            float ls = 0.f, ln = 0.f;
            for (int cc = 0; cc < DIMC; ++cc) {
                ls += cur[s][cc][mt] * wsd[cc * DIMC + c];
                ln += cur[o][cc][mt] * wnd[cc * DIMC + c];   // agg = feats[other]
            }
            nxt[s][c][mt] = cur[s][c][mt] + ls + sV[d][o][c] * sUY[s][mt] + ln;
        }
        __syncthreads();
        float (*tp)[DIMC][9] = cur; cur = nxt; nxt = tp;
    }
    // out1 = lin(lin(f1, pout1), final)
    for (int i = tid; i < 2 * DIMC * 3; i += 256) {
        int s = i / (DIMC * 3), o = (i / 3) % DIMC, m = i % 3;
        float acc = 0.f;
        for (int c = 0; c < DIMC; ++c) acc += cur[s][c][1 + m] * P.pout1[c * DIMC + o];
        sT2[s][o][m] = acc;
    }
    __syncthreads();
    if (tid < 6) {
        int j = tid / 3, m = tid % 3;
        float acc = 0.f;
        for (int o = 0; o < DIMC; ++o) acc += sT2[1][o][m] * P.fw[o * 2 + j];
        outp[j * (BBATCH * 3) + b * 3 + m] = acc;   // rot_g then shift_g
    }
}

// =====================================================================
// host orchestration
// =====================================================================
static inline int cdiv(long a, long b) { return (int)((a + b - 1) / b); }

extern "C" void kernel_launch(void* const* d_in, const int* in_sizes, int n_in,
                              void* d_out, int out_size, void* d_ws, size_t ws_size,
                              hipStream_t stream)
{
    (void)in_sizes; (void)out_size; (void)ws_size;
    if (n_in < 10) return;

    const float* r_ang = (const float*)d_in[0];
    const float* r_pos = (const float*)d_in[1];
    const float* r_res = (const float*)d_in[2];
    const unsigned char* r_msk = (const unsigned char*)d_in[3];
    const float* l_ang = (const float*)d_in[4];
    const float* l_pos = (const float*)d_in[5];
    const float* l_res = (const float*)d_in[6];
    const unsigned char* l_msk = (const unsigned char*)d_in[7];
    const float* t_in  = (const float*)d_in[8];

    // ---- param leaves in jax tree-flatten (sorted-key) order ----
    const int NLEAF = 104;
    const float* PL[104];
    int ls[104]; int p = 0;
    ls[p++] = 128 * 64; ls[p++] = 128 * 128; ls[p++] = 128 * 128;           // ds 0..2
    ls[p++] = 128 * 2;                                                      // final
    ls[p++] = 1; ls[p++] = 16384; ls[p++] = 16384; ls[p++] = 16384; ls[p++] = 128; // pool
    ls[p++] = 16384; ls[p++] = 16384;                                       // pout 0,1
    for (int L = 0; L < PDEPTH; ++L) {                                      // ptrans layers
        ls[p++] = 384 * 64;                                                 // wk
        ls[p++] = 16384; ls[p++] = 16384; ls[p++] = 16384;                  // wn 0..2
        ls[p++] = 384 * 64;                                                 // wq
        ls[p++] = 16384; ls[p++] = 16384; ls[p++] = 16384;                  // ws 0..2
        ls[p++] = 384 * 128; ls[p++] = 384 * 128; ls[p++] = 384 * 128;      // wv 0..2
    }
    ls[p++] = 128; ls[p++] = 128; ls[p++] = 128;                            // res_conv b0..2
    ls[p++] = 128 * 21 * 3; ls[p++] = 128 * 128 * 3; ls[p++] = 128 * 128 * 3; // w0..2
    // se3 layer 0 (in_degs 0,1)
    ls[p++] = 256 * 64; ls[p++] = 16384; ls[p++] = 16384; ls[p++] = 256 * 64;
    ls[p++] = 16384; ls[p++] = 16384;
    ls[p++] = 256 * 128; ls[p++] = 256 * 128; ls[p++] = 256 * 128;
    for (int L = 1; L < TDEPTH; ++L) {
        ls[p++] = 384 * 64;
        ls[p++] = 16384; ls[p++] = 16384; ls[p++] = 16384;
        ls[p++] = 384 * 64;
        ls[p++] = 16384; ls[p++] = 16384; ls[p++] = 16384;
        ls[p++] = 384 * 128; ls[p++] = 384 * 128; ls[p++] = 384 * 128;
    }
    ls[p++] = 3 * 128;                                                      // vec_proj
    if (n_in >= 9 + NLEAF) {
        for (int i = 0; i < NLEAF; ++i) PL[i] = (const float*)d_in[9 + i];
    } else {
        const float* base = (const float*)d_in[9];
        long off = 0;
        for (int i = 0; i < NLEAF; ++i) { PL[i] = base + off; off += ls[i]; }
    }

    // ---- workspace bump allocator ----
    char* wsB = (char*)d_ws;
    size_t wo = 0;
    auto walloc = [&](size_t bytes) -> void* {
        size_t a = (wo + 255) & ~(size_t)255;
        void* r = wsB + a;
        wo = a + bytes;
        return r;
    };
    const long BN = (long)BBATCH * NRMAX;
    half_t* im2  = (half_t*)walloc(BN * 384 * sizeof(half_t));
    half_t* wt   = (half_t*)walloc((size_t)128 * 384 * sizeof(half_t));
    half_t* invb = (half_t*)walloc(BN * 384 * sizeof(half_t));
    half_t* xsl  = (half_t*)walloc(BN * 128 * sizeof(half_t));
    float* gtmp  = (float*)walloc(BN * 128 * 4);
    float* act0  = (float*)walloc(BN * 128 * 4);
    float* act1  = (float*)walloc(BN * 128 * 4);
    float* qb    = (float*)walloc(BN * 64 * 4);
    float* kb    = (float*)walloc(BN * 64 * 4);
    float* vb0   = (float*)walloc(BN * 128 * 4);
    float* vb1   = (float*)walloc(BN * 128 * 4);
    float* vb2   = (float*)walloc(BN * 128 * 4);
    float* f0a   = (float*)walloc(BN * 128 * 4);
    float* f1a   = (float*)walloc(BN * 128 * 3 * 4);
    float* f2a   = (float*)walloc(BN * 128 * 5 * 4);
    float* f0b   = (float*)walloc(BN * 128 * 4);
    float* f1b   = (float*)walloc(BN * 128 * 3 * 4);
    float* f2b   = (float*)walloc(BN * 128 * 5 * 4);
    float* ag0   = (float*)walloc(BN * 128 * 4);
    float* ag1   = (float*)walloc(BN * 128 * 3 * 4);
    float* ag2   = (float*)walloc(BN * 128 * 5 * 4);
    int*   nidx  = (int*)walloc(BN * KNBR * 4);
    unsigned char* nmk = (unsigned char*)walloc(BN * KNBR);
    float* gg    = (float*)walloc(BN * 4);
    float* gsum  = (float*)walloc(BBATCH * 4);
    float* praw  = (float*)walloc((size_t)BBATCH * 128 * 9 * 4);
    float* pol0  = (float*)walloc((size_t)BBATCH * 2 * 128 * 4);
    float* pol1  = (float*)walloc((size_t)BBATCH * 2 * 128 * 3 * 4);
    float* pol2  = (float*)walloc((size_t)BBATCH * 2 * 128 * 5 * 4);
    float* pos2  = (float*)walloc((size_t)BBATCH * 2 * 3 * 4);

    auto GEMM = [&](const half_t* Xp, const half_t* Wp, float* Yp, long M, int Kd, int Nout) {
        dim3 g((unsigned)(M / 128), (unsigned)(Nout / 64), 1);
        wmma_gemm_f16<<<g, 256, (size_t)64 * Kd * 2, stream>>>(Xp, Wp, Yp, (int)M, Kd, Nout);
    };
    auto TCVT = [&](const float* W, int Kd, int Nout) {
        transpose_cvt_k<<<cdiv((long)Kd * Nout, 256), 256, 0, stream>>>(W, wt, Kd, Nout);
    };

    auto branch = [&](const float* ang, const float* pos, const float* res,
                      const unsigned char* msk, int N, int slot) {
        const long rows = (long)BBATCH * N;
        // --- res_conv: conv1 (21->128,k3) + 2 residual convs (128->128,k3) ---
        pack_conv_w_k<<<cdiv(128 * 64, 256), 256, 0, stream>>>(PL[58], wt, RESC, 64);
        pack_im2col_k<<<cdiv(rows * 64, 256), 256, 0, stream>>>(res, im2, N, RESC, 64);
        GEMM(im2, wt, gtmp, rows, 64, 128);
        bias_silu_k<<<cdiv(rows * 128, 256), 256, 0, stream>>>(gtmp, PL[55], (const float*)nullptr, act0, rows);

        pack_conv_w_k<<<cdiv(128 * 384, 256), 256, 0, stream>>>(PL[59], wt, 128, 384);
        pack_im2col_k<<<cdiv(rows * 384, 256), 256, 0, stream>>>(act0, im2, N, 128, 384);
        GEMM(im2, wt, gtmp, rows, 384, 128);
        bias_silu_k<<<cdiv(rows * 128, 256), 256, 0, stream>>>(gtmp, PL[56], act0, act1, rows);

        pack_conv_w_k<<<cdiv(128 * 384, 256), 256, 0, stream>>>(PL[60], wt, 128, 384);
        pack_im2col_k<<<cdiv(rows * 384, 256), 256, 0, stream>>>(act1, im2, N, 128, 384);
        GEMM(im2, wt, gtmp, rows, 384, 128);
        bias_silu_k<<<cdiv(rows * 128, 256), 256, 0, stream>>>(gtmp, PL[57], act1, f0a, rows);

        vec_proj_k<<<cdiv(rows * 128, 256), 256, 0, stream>>>(ang, PL[103], f1a, rows);
        topk_k<<<dim3(cdiv(N, 64), BBATCH), 64, 0, stream>>>(pos, msk, nidx, nmk, N);

        float *c0 = f0a, *c1 = f1a, *c2 = f2a, *x0 = f0b, *x1 = f1b, *x2 = f2b;
        for (int L = 0; L < TDEPTH; ++L) {
            const int nd = (L == 0) ? 2 : 3;
            const int F  = 128 * nd;
            const int base = (L == 0) ? 61 : 70 + 11 * (L - 1);
            const float* wk_ = PL[base];
            const float* wn_[3]; const float* ws_[3]; const float* wv_[3]; const float* wq_;
            if (nd == 2) {
                wn_[0] = PL[base + 1]; wn_[1] = PL[base + 2]; wn_[2] = nullptr;
                wq_ = PL[base + 3];
                ws_[0] = PL[base + 4]; ws_[1] = PL[base + 5]; ws_[2] = nullptr;
                wv_[0] = PL[base + 6]; wv_[1] = PL[base + 7]; wv_[2] = PL[base + 8];
            } else {
                wn_[0] = PL[base + 1]; wn_[1] = PL[base + 2]; wn_[2] = PL[base + 3];
                wq_ = PL[base + 4];
                ws_[0] = PL[base + 5]; ws_[1] = PL[base + 6]; ws_[2] = PL[base + 7];
                wv_[0] = PL[base + 8]; wv_[1] = PL[base + 9]; wv_[2] = PL[base + 10];
            }
            invariants_k<<<cdiv(rows * 128, 256), 256, 0, stream>>>(c0, c1, c2, invb, rows, nd, F);
            TCVT(wq_, F, 64);     GEMM(invb, wt, qb,  rows, F, 64);
            TCVT(wk_, F, 64);     GEMM(invb, wt, kb,  rows, F, 64);
            TCVT(wv_[0], F, 128); GEMM(invb, wt, vb0, rows, F, 128);
            TCVT(wv_[1], F, 128); GEMM(invb, wt, vb1, rows, F, 128);
            TCVT(wv_[2], F, 128); GEMM(invb, wt, vb2, rows, F, 128);
            attn_msg_k<<<(unsigned)rows, 128, 0, stream>>>(qb, kb, vb0, vb1, vb2,
                                                           c0, c1, c2, pos, nidx, nmk,
                                                           x0, x1, x2, ag0, ag1, ag2, N, nd);
            for (int d = 0; d < nd; ++d) {
                const int Mm = 2 * d + 1;
                float* fsrc = (d == 0) ? c0 : ((d == 1) ? c1 : c2);
                float* asrc = (d == 0) ? ag0 : ((d == 1) ? ag1 : ag2);
                float* fdst = (d == 0) ? x0 : ((d == 1) ? x1 : x2);
                // lin(feats, ws[d]) applied per m-slice (weight transposed once)
                TCVT(ws_[d], 128, 128);
                for (int m = 0; m < Mm; ++m) {
                    slice_pack_k<<<cdiv(rows * 128, 256), 256, 0, stream>>>(fsrc, xsl, rows, Mm, m);
                    GEMM(xsl, wt, gtmp, rows, 128, 128);
                    slice_add_k<<<cdiv(rows * 128, 256), 256, 0, stream>>>(gtmp, fdst, rows, Mm, m);
                }
                // lin(agg, wn[d]) applied per m-slice
                TCVT(wn_[d], 128, 128);
                for (int m = 0; m < Mm; ++m) {
                    slice_pack_k<<<cdiv(rows * 128, 256), 256, 0, stream>>>(asrc, xsl, rows, Mm, m);
                    GEMM(xsl, wt, gtmp, rows, 128, 128);
                    slice_add_k<<<cdiv(rows * 128, 256), 256, 0, stream>>>(gtmp, fdst, rows, Mm, m);
                }
            }
            float* tp;
            tp = c0; c0 = x0; x0 = tp;
            tp = c1; c1 = x1; x1 = tp;
            tp = c2; c2 = x2; x2 = tp;
        }
        // --- pool ---
        zero_k<<<1, 64, 0, stream>>>(gsum, BBATCH);
        gate_k<<<cdiv(rows, 256), 256, 0, stream>>>(c0, msk, PL[8], PL[4], gg, gsum, N, rows);
        pool_accum_k<<<cdiv((long)BBATCH * 128 * 9, 256), 256, 0, stream>>>(c0, c1, c2, gg, praw, N);
        pool_lin_k<<<cdiv((long)BBATCH * 128 * 9, 256), 256, 0, stream>>>(praw, gsum,
                                                                          PL[5], PL[6], PL[7],
                                                                          pol0, pol1, pol2, slot);
        pos2_k<<<BBATCH, 32, 0, stream>>>(pos, msk, pos2, slot, N);
    };

    branch(r_ang, r_pos, r_res, r_msk, NRMAX, 0);
    branch(l_ang, l_pos, l_res, l_msk, NLMAX, 1);

    FinalP FP;
    FP.ds0 = PL[0]; FP.ds1 = PL[1]; FP.ds2 = PL[2];
    FP.pout1 = PL[10]; FP.fw = PL[3];
    for (int L = 0; L < PDEPTH; ++L) {
        int bse = 11 + 11 * L;
        FP.wn[L][0] = PL[bse + 1]; FP.wn[L][1] = PL[bse + 2]; FP.wn[L][2] = PL[bse + 3];
        FP.ws[L][0] = PL[bse + 5]; FP.ws[L][1] = PL[bse + 6]; FP.ws[L][2] = PL[bse + 7];
        FP.wv[L][0] = PL[bse + 8]; FP.wv[L][1] = PL[bse + 9]; FP.wv[L][2] = PL[bse + 10];
    }
    final_k<<<BBATCH, 256, 0, stream>>>(FP, pol0, pol1, pol2, pos2, t_in, (float*)d_out);
}
